// StatefulMambaAudioProcessor_84722524881494
// MI455X (gfx1250) — compile-verified, matrix operand order checked by reading, not run on hardware
//
#include <hip/hip_runtime.h>
#include <hip/hip_bf16.h>

#define HID   512
#define NCB   4
#define VOC   1024
#define NST   64
#define NLAY  2
#define BATCH 4
#define TLEN  512
#define MROWS (BATCH * TLEN)   // 2048

typedef __attribute__((ext_vector_type(16))) __bf16 bfvec16;
typedef __attribute__((ext_vector_type(8)))  __bf16 bfvec8;
typedef __attribute__((ext_vector_type(8)))  float  fvec8;
typedef __attribute__((ext_vector_type(4)))  float  fvec4;

__device__ __forceinline__ float silu_f(float x) {
    return x / (1.0f + __expf(-x));
}

// ---------------------------------------------------------------------------
// Kernel 1: token embedding mean + positional embedding + broadcast context
// ---------------------------------------------------------------------------
__global__ __launch_bounds__(256) void k_embed(
    const int* __restrict__ tok,        // (B, NCB, T)
    const float* __restrict__ tctx,     // (B, HID)
    const float* __restrict__ temb,     // (NCB, VOC, HID)
    const float* __restrict__ pemb,     // (3000, HID)
    float* __restrict__ out)            // (M, HID)
{
    int m = blockIdx.x;
    int b = m / TLEN, t = m % TLEN;
    int i0 = tok[(b * NCB + 0) * TLEN + t];
    int i1 = tok[(b * NCB + 1) * TLEN + t];
    int i2 = tok[(b * NCB + 2) * TLEN + t];
    int i3 = tok[(b * NCB + 3) * TLEN + t];
    const float* e0 = temb + ((size_t)(0 * VOC) + i0) * HID;
    const float* e1 = temb + ((size_t)(1 * VOC) + i1) * HID;
    const float* e2 = temb + ((size_t)(2 * VOC) + i2) * HID;
    const float* e3 = temb + ((size_t)(3 * VOC) + i3) * HID;
    for (int h = threadIdx.x; h < HID; h += 256) {
        float a = 0.25f * (e0[h] + e1[h] + e2[h] + e3[h]);
        out[(size_t)m * HID + h] = a + pemb[(size_t)t * HID + h] + tctx[(size_t)b * HID + h];
    }
}

// ---------------------------------------------------------------------------
// Kernel 2: tiled WMMA GEMM   out = act(A @ W^T + bias)
//   A: (M,K) fp32 row-major (optional split A1[k<ksplit] | A2), W: (N,K) fp32.
//   Workgroup = 128 threads (4 waves) -> 64x64 output tile.
//   Software-pipelined: while 4x v_wmma_f32_16x16x32_bf16 consume LDS buffer
//   `cur`, the next K-chunk's 8x global_load_b128 are in flight and land
//   (as packed bf16) in buffer `1-cur`. One barrier per chunk.
// ---------------------------------------------------------------------------
#define LDSPITCH 40   // bf16 elements per LDS row (80B: 16B-aligned, bank-spread)

__device__ __forceinline__ bfvec16 ldsfrag(const __bf16* rowp, int kb) {
    bfvec8 lo = *(const bfvec8*)(rowp + kb);
    bfvec8 hi = *(const bfvec8*)(rowp + 16 + kb);
    return __builtin_shufflevector(lo, hi,
        0, 1, 2, 3, 4, 5, 6, 7, 8, 9, 10, 11, 12, 13, 14, 15);
}

__global__ __launch_bounds__(128) void k_wmma_gemm(
    const float* __restrict__ A1, int lda1,
    const float* __restrict__ A2, int lda2, int ksplit,
    const float* __restrict__ W,
    const float* __restrict__ bias,
    float* __restrict__ out, int ldc, int col_off,
    int K, int act, int head_mode)
{
    __shared__ alignas(16) __bf16 sA[2][64][LDSPITCH];
    __shared__ alignas(16) __bf16 sW[2][64][LDSPITCH];

    const int n0   = blockIdx.x * 64;
    const int m0   = blockIdx.y * 64;
    const int tid  = threadIdx.x;
    const int lane = tid & 31;
    const int wv   = tid >> 5;
    const int wm   = wv >> 1;          // wave's M half
    const int wn   = wv & 1;           // wave's N half
    const int r    = lane & 15;
    const int grp  = lane >> 4;
    const int kb   = grp * 8;

    fvec8 acc[2][2] = {};
    fvec4 va[4], vw[4];

    // register-batched loads of one 64x32 A tile + 64x32 W tile
    auto load_regs = [&](int kk) {
        const float* abase; int lda;
        if (A2 != nullptr && kk >= ksplit) {
            abase = A2 + (size_t)m0 * lda2 + (kk - ksplit); lda = lda2;
        } else {
            abase = A1 + (size_t)m0 * lda1 + kk;            lda = lda1;
        }
        const float* wbase = W + (size_t)n0 * K + kk;
#pragma unroll
        for (int i = 0; i < 4; ++i) {
            int q   = tid + i * 128;        // 0..511
            int row = q >> 3;
            int c4  = (q & 7) << 2;
            va[i] = *(const fvec4*)(abase + (size_t)row * lda + c4);
            vw[i] = *(const fvec4*)(wbase + (size_t)row * K   + c4);
        }
    };
    // convert fp32 -> bf16, store into LDS buffer
    auto store_lds = [&](int buf) {
#pragma unroll
        for (int i = 0; i < 4; ++i) {
            int q   = tid + i * 128;
            int row = q >> 3;
            int c4  = (q & 7) << 2;
#pragma unroll
            for (int j = 0; j < 4; ++j) {
                sA[buf][row][c4 + j] = (__bf16)va[i][j];
                sW[buf][row][c4 + j] = (__bf16)vw[i][j];
            }
        }
    };

    load_regs(0);
    store_lds(0);
    __syncthreads();

    int cur = 0;
    for (int kk = 0; kk < K; kk += 32) {
        const bool more = (kk + 32) < K;
        if (more) {
            load_regs(kk + 32);         // global loads overlap WMMA below
            // hint the chunk after that toward the caches (global_prefetch_b8)
            __builtin_prefetch(W + (size_t)(n0 + (tid >> 3)) * K + kk + 64, 0, 1);
        }

        bfvec16 a0 = ldsfrag(&sA[cur][wm * 32 + r][0],      kb);
        bfvec16 a1 = ldsfrag(&sA[cur][wm * 32 + 16 + r][0], kb);
        bfvec16 b0 = ldsfrag(&sW[cur][wn * 32 + r][0],      kb);
        bfvec16 b1 = ldsfrag(&sW[cur][wn * 32 + 16 + r][0], kb);

        acc[0][0] = __builtin_amdgcn_wmma_f32_16x16x32_bf16(
                        false, a0, false, b0, (short)0, acc[0][0], false, false);
        acc[0][1] = __builtin_amdgcn_wmma_f32_16x16x32_bf16(
                        false, a0, false, b1, (short)0, acc[0][1], false, false);
        acc[1][0] = __builtin_amdgcn_wmma_f32_16x16x32_bf16(
                        false, a1, false, b0, (short)0, acc[1][0], false, false);
        acc[1][1] = __builtin_amdgcn_wmma_f32_16x16x32_bf16(
                        false, a1, false, b1, (short)0, acc[1][1], false, false);

        if (more) store_lds(1 - cur);   // fill the other buffer
        __syncthreads();
        cur ^= 1;
    }

    // epilogue: bias / ReLU / optional bctv scatter
#pragma unroll
    for (int mi = 0; mi < 2; ++mi) {
#pragma unroll
        for (int ni = 0; ni < 2; ++ni) {
            const int n  = n0 + wn * 32 + ni * 16 + r;
            const float bv = bias ? bias[n] : 0.0f;
#pragma unroll
            for (int rr = 0; rr < 8; ++rr) {
                int m = m0 + wm * 32 + mi * 16 + (grp ? 8 + rr : rr);
                float v = acc[mi][ni][rr] + bv;
                if (act == 1) v = fmaxf(v, 0.0f);
                if (head_mode) {
                    int b  = m / TLEN, t  = m % TLEN;
                    int cc = n / VOC,  vv = n % VOC;
                    out[(((size_t)b * NCB + cc) * TLEN + t) * VOC + vv] = v;
                } else {
                    out[(size_t)m * ldc + col_off + n] = v;
                }
            }
        }
    }
}

// ---------------------------------------------------------------------------
// Kernel 3: depthwise causal conv (k=4, pad 3 left) + SiLU
// ---------------------------------------------------------------------------
__global__ __launch_bounds__(256) void k_dwconv_silu(
    const float* __restrict__ xz,   // (M, 2*HID); uses cols [0,HID)
    const float* __restrict__ cw,   // (HID, 1, 4)
    const float* __restrict__ cb,   // (HID,)
    float* __restrict__ xc)         // (M, HID)
{
    int idx = blockIdx.x * 256 + threadIdx.x;
    if (idx >= MROWS * HID) return;
    int d = idx & (HID - 1);
    int m = idx / HID;
    int t = m % TLEN;
    float acc = cb[d];
#pragma unroll
    for (int k = 0; k < 4; ++k) {
        int ts = t - 3 + k;
        if (ts >= 0)
            acc += xz[((size_t)(m - t + ts)) * (2 * HID) + d] * cw[d * 4 + k];
    }
    xc[(size_t)m * HID + d] = silu_f(acc);
}

// ---------------------------------------------------------------------------
// Kernel 4: sequential selective scan. 1 thread = 1 (batch, channel),
// 64 states + 64 decay factors in registers, B_t/C_t staged in LDS.
// Fuses the gating multiply by silu(x_res).
// ---------------------------------------------------------------------------
__global__ __launch_bounds__(256) void k_scan(
    const float* __restrict__ xc,     // (M, HID)
    const float* __restrict__ xz,     // (M, 2*HID) ; x_res = cols [HID,2HID)
    const float* __restrict__ bs,     // (M, NST)
    const float* __restrict__ cs,     // (M, NST)
    const float* __restrict__ A_log,  // (HID, NST)
    const float* __restrict__ dtp,    // (HID,)
    const float* __restrict__ Dp,     // (HID,)
    float* __restrict__ ys)           // (M, HID)
{
    const int b = blockIdx.x;
    const int d = blockIdx.y * 256 + threadIdx.x;

    const float dtd = logf(1.0f + __expf(dtp[d]));   // softplus
    const float Dd  = Dp[d];

    float dA[NST], h[NST];
#pragma unroll
    for (int n = 0; n < NST; ++n) {
        dA[n] = 1.0f + dtd * (-__expf(A_log[(size_t)d * NST + n]));
        h[n]  = 0.0f;
    }

    __shared__ float sB[NST];
    __shared__ float sC[NST];

    for (int t = 0; t < TLEN; ++t) {
        const size_t row = (size_t)b * TLEN + t;
        __syncthreads();
        if (threadIdx.x < NST)            sB[threadIdx.x]       = bs[row * NST + threadIdx.x];
        else if (threadIdx.x < 2 * NST)   sC[threadIdx.x - NST] = cs[row * NST + threadIdx.x - NST];
        __syncthreads();

        const float u   = xc[row * HID + d];
        const float xr  = xz[row * (2 * HID) + HID + d];
        const float dtu = dtd * u;
        float accv = 0.0f;
#pragma unroll
        for (int n = 0; n < NST; ++n) {
            h[n] = h[n] * dA[n] + dtu * sB[n];
            accv += sC[n] * h[n];
        }
        float y = accv + Dd * u;
        ys[row * HID + d] = y * silu_f(xr);
    }
}

// ---------------------------------------------------------------------------
// Kernel 5: row LayerNorm over HID=512 (optional residual input)
// ---------------------------------------------------------------------------
__global__ __launch_bounds__(256) void k_ln(
    const float* __restrict__ x,
    const float* __restrict__ res,    // nullable
    const float* __restrict__ g,
    const float* __restrict__ beta,
    float* __restrict__ out, int ldo, int coff)
{
    const int m   = blockIdx.x;
    const int tid = threadIdx.x;
    float v0 = x[(size_t)m * HID + tid];
    float v1 = x[(size_t)m * HID + tid + 256];
    if (res) {
        v0 += res[(size_t)m * HID + tid];
        v1 += res[(size_t)m * HID + tid + 256];
    }
    __shared__ float sred[256];
    sred[tid] = v0 + v1;
    __syncthreads();
    for (int s = 128; s > 0; s >>= 1) {
        if (tid < s) sred[tid] += sred[tid + s];
        __syncthreads();
    }
    const float mean = sred[0] * (1.0f / HID);
    const float d0 = v0 - mean, d1 = v1 - mean;
    __syncthreads();
    sred[tid] = d0 * d0 + d1 * d1;
    __syncthreads();
    for (int s = 128; s > 0; s >>= 1) {
        if (tid < s) sred[tid] += sred[tid + s];
        __syncthreads();
    }
    const float inv = rsqrtf(sred[0] * (1.0f / HID) + 1e-5f);
    out[(size_t)m * ldo + coff + tid]       = d0 * inv * g[tid]       + beta[tid];
    out[(size_t)m * ldo + coff + tid + 256] = d1 * inv * g[tid + 256] + beta[tid + 256];
}

// ---------------------------------------------------------------------------
// Kernel 6: 4 dilated 1D context convs (HID -> 128 each, k=3, pad=dil).
// Block = (row m, dilation i); 3 input taps staged in LDS.
// ---------------------------------------------------------------------------
__global__ __launch_bounds__(128) void k_ctxconv(
    const float* __restrict__ emb,
    const float* __restrict__ w0, const float* __restrict__ w1,
    const float* __restrict__ w2, const float* __restrict__ w3,
    const float* __restrict__ b0, const float* __restrict__ b1,
    const float* __restrict__ b2, const float* __restrict__ b3,
    float* __restrict__ ctx)           // (M, HID)
{
    const int m = blockIdx.x;
    const int i = blockIdx.y;
    const int b = m / TLEN, t = m % TLEN;
    const int dils[4] = {1, 3, 9, 27};
    const int dil = dils[i];
    const float* w  = (i == 0) ? w0 : (i == 1) ? w1 : (i == 2) ? w2 : w3;
    const float* bb = (i == 0) ? b0 : (i == 1) ? b1 : (i == 2) ? b2 : b3;

    __shared__ float sx[3][HID];
#pragma unroll
    for (int k = 0; k < 3; ++k) {
        int ts = t + (k - 1) * dil;
        bool ok = (ts >= 0) && (ts < TLEN);
        const float* src = emb + (size_t)(b * TLEN + ts) * HID;
        for (int hh = threadIdx.x; hh < HID; hh += 128)
            sx[k][hh] = ok ? src[hh] : 0.0f;
    }
    __syncthreads();

    const int o = threadIdx.x;
    float acc = bb[o];
    const float* wo = w + (size_t)o * HID * 3;
    for (int hh = 0; hh < HID; ++hh) {
        acc += sx[0][hh] * wo[hh * 3 + 0]
             + sx[1][hh] * wo[hh * 3 + 1]
             + sx[2][hh] * wo[hh * 3 + 2];
    }
    ctx[(size_t)m * HID + i * 128 + o] = acc;
}

// ---------------------------------------------------------------------------
// Host launcher
// ---------------------------------------------------------------------------
extern "C" void kernel_launch(void* const* d_in, const int* in_sizes, int n_in,
                              void* d_out, int out_size, void* d_ws, size_t ws_size,
                              hipStream_t stream) {
    (void)in_sizes; (void)n_in; (void)out_size; (void)ws_size;

    const int*   tok    = (const int*)  d_in[0];
    const float* tctx   = (const float*)d_in[1];
    const float* temb   = (const float*)d_in[2];
    const float* pemb   = (const float*)d_in[3];
    // layers: 11 tensors each, starting at index 4
    const float* ctx_w[4] = {(const float*)d_in[26], (const float*)d_in[27],
                             (const float*)d_in[28], (const float*)d_in[29]};
    const float* ctx_b[4] = {(const float*)d_in[30], (const float*)d_in[31],
                             (const float*)d_in[32], (const float*)d_in[33]};
    const float* fusion_W    = (const float*)d_in[34];
    const float* fusion_b    = (const float*)d_in[35];
    const float* fusion_g    = (const float*)d_in[36];
    const float* fusion_beta = (const float*)d_in[37];
    const float* head_W      = (const float*)d_in[38];
    const float* head_b      = (const float*)d_in[39];
    float* out = (float*)d_out;

    // workspace layout (floats)
    float* ws    = (float*)d_ws;
    float* emb   = ws;                          // M*HID
    float* xz    = emb  + (size_t)MROWS * HID;  // M*2HID
    float* xcb   = xz   + (size_t)MROWS * 2 * HID;
    float* bsb   = xcb  + (size_t)MROWS * HID;  // M*NST
    float* csb   = bsb  + (size_t)MROWS * NST;
    float* ysb   = csb  + (size_t)MROWS * NST;  // M*HID
    float* yob   = ysb  + (size_t)MROWS * HID;
    float* fA    = yob  + (size_t)MROWS * HID;
    float* fB    = fA   + (size_t)MROWS * HID;
    float* ctxb  = fB   + (size_t)MROWS * HID;
    float* fus   = ctxb + (size_t)MROWS * HID;
    float* fln   = fus  + (size_t)MROWS * HID;

    // 1. embedding
    k_embed<<<MROWS, 256, 0, stream>>>(tok, tctx, temb, pemb, emb);

    // 2. SSM layers
    const float* feats_in = emb;
    float* feats_out = fA;
    for (int l = 0; l < NLAY; ++l) {
        const int base = 4 + l * 11;
        const float* in_W   = (const float*)d_in[base + 0];
        const float* conv_w = (const float*)d_in[base + 1];
        const float* conv_b = (const float*)d_in[base + 2];
        const float* A_log  = (const float*)d_in[base + 3];
        const float* dt     = (const float*)d_in[base + 4];
        const float* Dvec   = (const float*)d_in[base + 5];
        const float* B_W    = (const float*)d_in[base + 6];
        const float* C_W    = (const float*)d_in[base + 7];
        const float* out_W  = (const float*)d_in[base + 8];
        const float* ln_g   = (const float*)d_in[base + 9];
        const float* ln_b   = (const float*)d_in[base + 10];

        // xz = feats_in @ in_W^T   (M x 1024)
        k_wmma_gemm<<<dim3(2 * HID / 64, MROWS / 64), 128, 0, stream>>>(
            feats_in, HID, nullptr, 0, 0, in_W, nullptr, xz, 2 * HID, 0,
            HID, 0, 0);
        // depthwise conv + silu
        k_dwconv_silu<<<(MROWS * HID + 255) / 256, 256, 0, stream>>>(xz, conv_w, conv_b, xcb);
        // B/C projections (M x 64)
        k_wmma_gemm<<<dim3(NST / 64, MROWS / 64), 128, 0, stream>>>(
            xcb, HID, nullptr, 0, 0, B_W, nullptr, bsb, NST, 0, HID, 0, 0);
        k_wmma_gemm<<<dim3(NST / 64, MROWS / 64), 128, 0, stream>>>(
            xcb, HID, nullptr, 0, 0, C_W, nullptr, csb, NST, 0, HID, 0, 0);
        // sequential scan (fuses * silu(x_res))
        k_scan<<<dim3(BATCH, HID / 256), 256, 0, stream>>>(
            xcb, xz, bsb, csb, A_log, dt, Dvec, ysb);
        // out projection
        k_wmma_gemm<<<dim3(HID / 64, MROWS / 64), 128, 0, stream>>>(
            ysb, HID, nullptr, 0, 0, out_W, nullptr, yob, HID, 0, HID, 0, 0);
        // LN(residual)
        k_ln<<<MROWS, 256, 0, stream>>>(yob, feats_in, ln_g, ln_b, feats_out, HID, 0);

        feats_in  = feats_out;
        feats_out = (l == 0) ? fB : fA;
    }

    // 3. dilated context convs on audio_emb
    k_ctxconv<<<dim3(MROWS, 4), 128, 0, stream>>>(
        emb, ctx_w[0], ctx_w[1], ctx_w[2], ctx_w[3],
        ctx_b[0], ctx_b[1], ctx_b[2], ctx_b[3], ctxb);

    // 4. fusion GEMM over concat [feats | ctx]  (K=1024) + bias + ReLU
    k_wmma_gemm<<<dim3(HID / 64, MROWS / 64), 128, 0, stream>>>(
        feats_in, HID, ctxb, HID, HID, fusion_W, fusion_b, fus, HID, 0,
        2 * HID, 1, 0);

    // 5. fusion LayerNorm
    k_ln<<<MROWS, 256, 0, stream>>>(fus, nullptr, fusion_g, fusion_beta, fln, HID, 0);

    // 6. head GEMM -> (B, NCB, T, VOC) logits, bias fused, scatter epilogue
    k_wmma_gemm<<<dim3(NCB * VOC / 64, MROWS / 64), 128, 0, stream>>>(
        fln, HID, nullptr, 0, 0, head_W, head_b, out, 0, 0, HID, 0, 1);
}
